// UnifiedAttention_51857435131905
// MI455X (gfx1250) — compile-verified, matrix operand order checked by reading, not run on hardware
//
#include <hip/hip_runtime.h>
#include <hip/hip_bf16.h>
#include <stdint.h>

typedef __attribute__((ext_vector_type(16))) __bf16 v16bf;
typedef __attribute__((ext_vector_type(8)))  __bf16 v8bf;
typedef __attribute__((ext_vector_type(4)))  __bf16 v4bf;
typedef __attribute__((ext_vector_type(8)))  float  v8f;

#define S_LEN 2048
#define DIM   128
#define HDIM  64
#define NHEAD 16
#define NKVH  4
#define REP   4
#define BM    128   // query rows per workgroup
#define BN    32    // keys per iteration
#define NWAVE 8
#define KSTR  136   // sK row stride (elems): 272B -> bank-conflict-free b128 frag reads
#define VSTR  56    // sVT / sP row stride (elems): 112B -> conflict-free b128 frag reads

__device__ __forceinline__ __bf16 to_bf16(float f) {
  union { float f; uint32_t u; } v; v.f = f;
  uint32_t r = v.u + 0x7FFFu + ((v.u >> 16) & 1u);   // round-to-nearest-even
  uint16_t h = (uint16_t)(r >> 16);
  __bf16 b;
  __builtin_memcpy(&b, &h, 2);
  return b;
}

__device__ __forceinline__ v16bf frag_cat(v8bf lo, v8bf hi) {
  return __builtin_shufflevector(lo, hi, 0, 1, 2, 3, 4, 5, 6, 7,
                                 8, 9, 10, 11, 12, 13, 14, 15);
}

// one aligned 16B LDS vector load (8 bf16)
__device__ __forceinline__ v8bf ld8(const __bf16* p) {
  return *(const v8bf*)p;
}

// 8 consecutive floats from global (pre-scaled) -> 8 bf16
__device__ __forceinline__ v8bf cvt8g(const float* p, float s) {
  float4 a = *(const float4*)(p);
  float4 b = *(const float4*)(p + 4);
  v8bf r;
  r[0] = to_bf16(a.x * s); r[1] = to_bf16(a.y * s); r[2] = to_bf16(a.z * s); r[3] = to_bf16(a.w * s);
  r[4] = to_bf16(b.x * s); r[5] = to_bf16(b.y * s); r[6] = to_bf16(b.z * s); r[7] = to_bf16(b.w * s);
  return r;
}

__global__ __launch_bounds__(256)
void diffattn_wmma_kernel(const float* __restrict__ Q, const float* __restrict__ K,
                          const float* __restrict__ V, float* __restrict__ O) {
  __shared__ __align__(16) __bf16 sK [BN][KSTR];           // [key][feature], row-major K
  __shared__ __align__(16) __bf16 sVT[DIM + 16][VSTR];     // [dim][key]; rows DIM.. = ones-column tile
  __shared__ __align__(16) __bf16 sP1[NWAVE][16][VSTR];    // per-wave P (stream 1)
  __shared__ __align__(16) __bf16 sP2[NWAVE][16][VSTR];    // per-wave P (stream 2)

  const int tid    = threadIdx.x;
  const int lane   = tid & 31;
  const int wave   = tid >> 5;
  const int laneHi = lane >> 4;     // 0/1
  const int laneLo = lane & 15;
  const int koff   = laneHi << 3;   // per-lane-half K offset within fragment
  const int head   = blockIdx.y;
  const int kvh    = head / REP;
  const int qrow0  = blockIdx.x * BM + wave * 16;
  const int rowLast = qrow0 + 15;
  const int kmax   = (blockIdx.x + 1) * BM;       // causal upper bound for this block

  const float scale = 0.125f;       // 1/sqrt(64), folded into Q fragments
  const float lam   = 0.8f;
  const float normf = 0.2f;         // 1 - lambda_init

  // ones-column extra V tile (dim DIM -> 1.0, DIM+1..DIM+15 -> 0), written once
  if (tid < 16) {
    __bf16 fill = (tid == 0) ? to_bf16(1.0f) : to_bf16(0.0f);
    for (int j = 0; j < BN; ++j) sVT[DIM + tid][j] = fill;
  }

  // ---- Q fragments (A layout), pre-scaled by 1/sqrt(64) ----
  v16bf q1f[2], q2f[2];
  {
    const float* qp = Q + ((size_t)head * S_LEN + (size_t)(qrow0 + laneLo)) * DIM;
#pragma unroll
    for (int c = 0; c < 2; ++c) {
      int k0 = c * 32 + koff;
      q1f[c] = frag_cat(cvt8g(qp + k0, scale),        cvt8g(qp + k0 + 16, scale));
      q2f[c] = frag_cat(cvt8g(qp + HDIM + k0, scale), cvt8g(qp + HDIM + k0 + 16, scale));
    }
  }

  v8f acc1[8], acc2[8], acc1x, acc2x;
  v8f vzero = {};
#pragma unroll
  for (int n = 0; n < 8; ++n) { acc1[n] = vzero; acc2[n] = vzero; }
  acc1x = vzero; acc2x = vzero;     // col 0 accumulates the softmax denominator
  float m1[8], m2[8];
#pragma unroll
  for (int i = 0; i < 8; ++i) { m1[i] = -3.0e38f; m2[i] = -3.0e38f; }

  for (int kb = 0; kb < kmax; kb += BN) {
    __syncthreads();
    // cooperative stage: K row-major (vector b64 stores), V transposed
    for (int idx = tid; idx < (BN * DIM) / 4; idx += 256) {
      int key  = idx >> 5;          // DIM/4 = 32 float4 per key
      int feat = (idx & 31) << 2;
      size_t g = ((size_t)kvh * S_LEN + (size_t)(kb + key)) * DIM + feat;
      float4 kv = *(const float4*)&K[g];
      float4 vv = *(const float4*)&V[g];
      v4bf kk;
      kk[0] = to_bf16(kv.x); kk[1] = to_bf16(kv.y);
      kk[2] = to_bf16(kv.z); kk[3] = to_bf16(kv.w);
      *(v4bf*)&sK[key][feat] = kk;
      sVT[feat + 0][key] = to_bf16(vv.x);
      sVT[feat + 1][key] = to_bf16(vv.y);
      sVT[feat + 2][key] = to_bf16(vv.z);
      sVT[feat + 3][key] = to_bf16(vv.w);
    }
    if (kb + BN < kmax) {   // CDNA5 global_prefetch of next tile
      __builtin_prefetch(&K[((size_t)kvh * S_LEN + (size_t)(kb + BN + lane)) * DIM], 0, 1);
      __builtin_prefetch(&V[((size_t)kvh * S_LEN + (size_t)(kb + BN + lane)) * DIM], 0, 1);
    }
    __syncthreads();

    if (kb <= rowLast) {    // wave-uniform skip of fully-masked tiles
      // ---- scores: S1 = (Q1*scale)*K1^T, S2 = (Q2*scale)*K2^T  (16x32 tile, K=64) ----
      v8f s1[2], s2[2];
#pragma unroll
      for (int nt = 0; nt < 2; ++nt) {
        const int key = nt * 16 + laneLo;
        v8f c1 = vzero, c2 = vzero;
#pragma unroll
        for (int c = 0; c < 2; ++c) {
          const int k0 = c * 32 + koff;
          v16bf b1 = frag_cat(ld8(&sK[key][k0]),        ld8(&sK[key][k0 + 16]));
          v16bf b2 = frag_cat(ld8(&sK[key][HDIM + k0]), ld8(&sK[key][HDIM + k0 + 16]));
          c1 = __builtin_amdgcn_wmma_f32_16x16x32_bf16(false, q1f[c], false, b1,
                                                       (short)0, c1, false, false);
          c2 = __builtin_amdgcn_wmma_f32_16x16x32_bf16(false, q2f[c], false, b2,
                                                       (short)0, c2, false, false);
        }
        s1[nt] = c1; s2[nt] = c2;
      }
      // causal mask: only diagonal tiles (kb+BN-1 > qrow0) contain masked elements
      if (kb + BN - 1 > qrow0) {
#pragma unroll
        for (int nt = 0; nt < 2; ++nt) {
#pragma unroll
          for (int i = 0; i < 8; ++i) {
            int row = qrow0 + i + (laneHi << 3);
            int col = kb + nt * 16 + laneLo;
            if (col > row) { s1[nt][i] = -1.0e9f; s2[nt][i] = -1.0e9f; }
          }
        }
      }
      // ---- dual-stream online softmax: max reduction only (sum comes from WMMA) ----
#pragma unroll
      for (int i = 0; i < 8; ++i) {
        float t = fmaxf(s1[0][i], s1[1][i]);
#pragma unroll
        for (int off = 8; off >= 1; off >>= 1) t = fmaxf(t, __shfl_xor(t, off));
        float mn = fmaxf(m1[i], t);
        float corr = __expf(m1[i] - mn);
        m1[i] = mn;
        s1[0][i] = __expf(s1[0][i] - mn);
        s1[1][i] = __expf(s1[1][i] - mn);
        acc1x[i] *= corr;
#pragma unroll
        for (int n = 0; n < 8; ++n) acc1[n][i] *= corr;
      }
#pragma unroll
      for (int i = 0; i < 8; ++i) {
        float t = fmaxf(s2[0][i], s2[1][i]);
#pragma unroll
        for (int off = 8; off >= 1; off >>= 1) t = fmaxf(t, __shfl_xor(t, off));
        float mn = fmaxf(m2[i], t);
        float corr = __expf(m2[i] - mn);
        m2[i] = mn;
        s2[0][i] = __expf(s2[0][i] - mn);
        s2[1][i] = __expf(s2[1][i] - mn);
        acc2x[i] *= corr;
#pragma unroll
        for (int n = 0; n < 8; ++n) acc2[n][i] *= corr;
      }
      // ---- P through LDS: C-layout -> A-layout (wave-private region) ----
#pragma unroll
      for (int nt = 0; nt < 2; ++nt) {
#pragma unroll
        for (int i = 0; i < 8; ++i) {
          int r = i + (laneHi << 3);
          int cidx = nt * 16 + laneLo;
          sP1[wave][r][cidx] = to_bf16(s1[nt][i]);
          sP2[wave][r][cidx] = to_bf16(s2[nt][i]);
        }
      }
      asm volatile("s_wait_dscnt 0x0" ::: "memory");
      v16bf p1f = frag_cat(ld8(&sP1[wave][laneLo][koff]), ld8(&sP1[wave][laneLo][koff + 16]));
      v16bf p2f = frag_cat(ld8(&sP2[wave][laneLo][koff]), ld8(&sP2[wave][laneLo][koff + 16]));
      // ---- O1 += P1*V, O2 += P2*V; V B-fragments shared across streams ----
#pragma unroll
      for (int nt = 0; nt < 8; ++nt) {
        const int dim = nt * 16 + laneLo;
        v16bf vb = frag_cat(ld8(&sVT[dim][koff]), ld8(&sVT[dim][koff + 16]));
        acc1[nt] = __builtin_amdgcn_wmma_f32_16x16x32_bf16(false, p1f, false, vb,
                                                           (short)0, acc1[nt], false, false);
        acc2[nt] = __builtin_amdgcn_wmma_f32_16x16x32_bf16(false, p2f, false, vb,
                                                           (short)0, acc2[nt], false, false);
      }
      // ones-column tile: accumulates softmax denominators into col 0 of accx
      {
        v16bf vbx = frag_cat(ld8(&sVT[DIM + laneLo][koff]), ld8(&sVT[DIM + laneLo][koff + 16]));
        acc1x = __builtin_amdgcn_wmma_f32_16x16x32_bf16(false, p1f, false, vbx,
                                                        (short)0, acc1x, false, false);
        acc2x = __builtin_amdgcn_wmma_f32_16x16x32_bf16(false, p2f, false, vbx,
                                                        (short)0, acc2x, false, false);
      }
    }
  }

  // ---- epilogue: l from col-0 lanes of accx; 16 divides; then mul/sub ----
  float r1[8], r2[8];
#pragma unroll
  for (int i = 0; i < 8; ++i) {
    float l1i = __shfl(acc1x[i], laneHi << 4);   // col 0 holds sum(P) per row
    float l2i = __shfl(acc2x[i], laneHi << 4);
    r1[i] = normf / l1i;
    r2[i] = (normf * lam) / l2i;
  }
  const size_t obase = (size_t)head * S_LEN * DIM;
#pragma unroll
  for (int nt = 0; nt < 8; ++nt) {
#pragma unroll
    for (int i = 0; i < 8; ++i) {
      int row = qrow0 + i + (laneHi << 3);
      int col = nt * 16 + laneLo;
      float o = acc1[nt][i] * r1[i] - acc2[nt][i] * r2[i];
      O[obase + (size_t)row * DIM + col] = o;
    }
  }
}

extern "C" void kernel_launch(void* const* d_in, const int* in_sizes, int n_in,
                              void* d_out, int out_size, void* d_ws, size_t ws_size,
                              hipStream_t stream) {
  (void)in_sizes; (void)n_in; (void)out_size; (void)d_ws; (void)ws_size;
  const float* Q = (const float*)d_in[0];
  const float* K = (const float*)d_in[1];
  const float* V = (const float*)d_in[2];
  float* O = (float*)d_out;
  dim3 grid(S_LEN / BM, NHEAD, 1);
  dim3 block(256, 1, 1);
  hipLaunchKernelGGL(diffattn_wmma_kernel, grid, block, 0, stream, Q, K, V, O);
}
